// LSTMCellUnknownData_77309411328201
// MI455X (gfx1250) — matrix operand order repaired and checked by reading, not compiled, not run on hardware
//
#include <hip/hip_runtime.h>

// ---------------------------------------------------------------------------
// cy2 = f2*cx2 + i2*z ; only nets k in {1,3,5} of the 7 are live.
// bf16 WMMA GEMMs (f32 accumulate), 64x64 wave tiles (16 WMMA per 8 frag
// loads -> 0.5KB LDS per WMMA), single-barrier double-buffered LDS staging.
// ---------------------------------------------------------------------------

typedef __bf16 bf16_t;
typedef bf16_t v16bf __attribute__((ext_vector_type(16)));
typedef bf16_t v4bf  __attribute__((ext_vector_type(4)));
typedef float  v8f   __attribute__((ext_vector_type(8)));

#define BATCH 16384
#define DIN   512
#define HDIM  512

#define WGM   128        // workgroup M tile
#define WGN   256        // workgroup N tile
#define KT    32         // K per step
#define LDSTR 40         // LDS row stride in bf16 (80B: 16B-aligned, bank-spread)

union FragBF { v16bf v; uint4 q[2]; };

__device__ __forceinline__ float sigmoidf_(float x) {
    return 1.0f / (1.0f + __expf(-x));
}

// ---------------------------------------------------------------------------
// fp32 -> bf16 pack, 8 elements per thread (n must be a multiple of 2048)
// ---------------------------------------------------------------------------
__global__ void cvt_f32_bf16(const float* __restrict__ src,
                             bf16_t* __restrict__ dst, int n) {
    int i = (blockIdx.x * blockDim.x + threadIdx.x) * 8;
    if (i >= n) return;
    float4 a = *(const float4*)(src + i);
    float4 b = *(const float4*)(src + i + 4);
    bf16_t t[8] = {(bf16_t)a.x, (bf16_t)a.y, (bf16_t)a.z, (bf16_t)a.w,
                   (bf16_t)b.x, (bf16_t)b.y, (bf16_t)b.z, (bf16_t)b.w};
    *(uint4*)(dst + i) = *(const uint4*)t;
}

// ---------------------------------------------------------------------------
// Core tile GEMM: D(128x256) += A(128x512) * B(512x256)^T-ish, bf16->f32.
// A row-major [*,512] contiguous in K; B rows contiguous in K ([n][k]).
// Each of 8 waves (2M x 4N) computes a 64x64 tile = 4x4 WMMA frags.
// Epilogue is a functor-style template parameter via enum.
// ---------------------------------------------------------------------------
template <int EPILOGUE>   // 0: tanh->bf16 (a1), 1: sigmoid(tanh)->bf16 (gate)
__device__ __forceinline__ void gemm_tile(
    const bf16_t* __restrict__ A,     // [BATCH][512]
    const bf16_t* __restrict__ Bw,    // [512][512] rows contiguous in K
    const float*  __restrict__ bias,  // [512]
    bf16_t* __restrict__ Dst,         // [BATCH][512] bf16
    bf16_t* As, bf16_t* Bs)           // LDS: As[2][128*LDSTR], Bs[2][256*LDSTR]
{
    const int tid   = threadIdx.x;
    const int lane  = tid & 31;
    const int wave  = tid >> 5;
    const int waveM = wave & 1;       // 0..1
    const int waveN = wave >> 1;      // 0..3
    const int lo    = lane & 15;
    const int hi    = lane >> 4;

    const int m0 = blockIdx.x * WGM;
    const int n0 = blockIdx.y * WGN;

    const int arow = tid >> 1, ahalf = tid & 1;   // A stage: 128 rows x 32 k
    const int bcol = tid;                          // B stage: 256 cols x 32 k

    const int ASZ = WGM * LDSTR;  // elements per A buffer
    const int BSZ = WGN * LDSTR;  // elements per B buffer

    v8f acc[4][4] = {};

    // ---- prologue: stage k-step 0 into buffer 0 ----
    {
        const bf16_t* gp = A + (size_t)(m0 + arow) * 512 + ahalf * 16;
        uint4 x0 = ((const uint4*)gp)[0];
        uint4 x1 = ((const uint4*)gp)[1];
        *(uint4*)&As[arow * LDSTR + ahalf * 16]     = x0;
        *(uint4*)&As[arow * LDSTR + ahalf * 16 + 8] = x1;
        const bf16_t* wp = Bw + (size_t)(n0 + bcol) * 512;
#pragma unroll
        for (int c = 0; c < 4; ++c) {
            uint4 y = ((const uint4*)wp)[c];
            *(uint4*)&Bs[bcol * LDSTR + c * 8] = y;
        }
    }
    __syncthreads();

    const int NK = 512 / KT;   // 16
    for (int ks = 0; ks < NK; ++ks) {
        const int cur = ks & 1;
        const bool more = (ks + 1) < NK;

        // ---- issue global loads for next k-step (latency hides under WMMA) --
        uint4 ax0, ax1, bx[4];
        if (more) {
            const int kn0 = (ks + 1) * KT;
            const bf16_t* gp = A + (size_t)(m0 + arow) * 512 + kn0 + ahalf * 16;
            __builtin_prefetch(gp + 32, 0, 1);
            ax0 = ((const uint4*)gp)[0];
            ax1 = ((const uint4*)gp)[1];
            const bf16_t* wp = Bw + (size_t)(n0 + bcol) * 512 + kn0;
#pragma unroll
            for (int c = 0; c < 4; ++c) bx[c] = ((const uint4*)wp)[c];
        }

        // ---- compute from current buffer: 4 A frags, 4 B frags, 16 WMMA ----
        const bf16_t* Ab = As + cur * ASZ;
        const bf16_t* Bb = Bs + cur * BSZ;
        FragBF af[4];
#pragma unroll
        for (int mi = 0; mi < 4; ++mi) {
            int row = waveM * 64 + mi * 16 + lo;
            af[mi].q[0] = *(const uint4*)&Ab[row * LDSTR + hi * 8];
            af[mi].q[1] = *(const uint4*)&Ab[row * LDSTR + 16 + hi * 8];
        }
#pragma unroll
        for (int ni = 0; ni < 4; ++ni) {
            int col = waveN * 64 + ni * 16 + lo;
            FragBF bfr;
            bfr.q[0] = *(const uint4*)&Bb[col * LDSTR + hi * 8];
            bfr.q[1] = *(const uint4*)&Bb[col * LDSTR + 16 + hi * 8];
#pragma unroll
            for (int mi = 0; mi < 4; ++mi) {
                acc[mi][ni] = __builtin_amdgcn_wmma_f32_16x16x32_bf16(
                    false, af[mi].v, false, bfr.v, (short)0, acc[mi][ni],
                    false, false);
            }
        }

        // ---- store staged regs into the other buffer (no read conflict) ----
        if (more) {
            bf16_t* An = As + (cur ^ 1) * ASZ;
            bf16_t* Bn = Bs + (cur ^ 1) * BSZ;
            *(uint4*)&An[arow * LDSTR + ahalf * 16]     = ax0;
            *(uint4*)&An[arow * LDSTR + ahalf * 16 + 8] = ax1;
#pragma unroll
            for (int c = 0; c < 4; ++c)
                *(uint4*)&Bn[bcol * LDSTR + c * 8] = bx[c];
        }
        __syncthreads();
    }

    // ---- epilogue ----
#pragma unroll
    for (int ni = 0; ni < 4; ++ni) {
        int n = n0 + waveN * 64 + ni * 16 + lo;
        float bv = bias[n];
#pragma unroll
        for (int mi = 0; mi < 4; ++mi) {
            int mb = m0 + waveM * 64 + mi * 16 + hi * 8;
#pragma unroll
            for (int r = 0; r < 8; ++r) {
                float h = acc[mi][ni][r] + bv;
                float v = (EPILOGUE == 0) ? tanhf(h)
                                          : sigmoidf_(tanhf(h));
                Dst[(size_t)(mb + r) * 512 + n] = (bf16_t)v;
            }
        }
    }
}

__launch_bounds__(256)
__global__ void gemm1_tanh(const bf16_t* __restrict__ hxb,
                           const bf16_t* __restrict__ w1b,   // [3][H][DIN]
                           const float*  __restrict__ b1,    // [7][H]
                           bf16_t* __restrict__ a1)          // [3][B][H]
{
    __shared__ __align__(16) bf16_t As[2 * WGM * LDSTR];
    __shared__ __align__(16) bf16_t Bs[2 * WGN * LDSTR];
    const int kn  = blockIdx.z;
    const int src = 2 * kn + 1;        // live nets: 1, 3, 5
    gemm_tile<0>(hxb,
                 w1b + (size_t)kn * HDIM * DIN,
                 b1 + (size_t)src * HDIM,
                 a1 + (size_t)kn * BATCH * HDIM,
                 As, Bs);
}

__launch_bounds__(256)
__global__ void gemm2_gate(const bf16_t* __restrict__ a1,    // [3][B][H]
                           const bf16_t* __restrict__ w2b,   // [3][H][H]
                           const float*  __restrict__ b2,    // [7][H]
                           bf16_t* __restrict__ gates)       // [3][B][H]
{
    __shared__ __align__(16) bf16_t As[2 * WGM * LDSTR];
    __shared__ __align__(16) bf16_t Bs[2 * WGN * LDSTR];
    const int kn  = blockIdx.z;
    const int src = 2 * kn + 1;
    gemm_tile<1>(a1 + (size_t)kn * BATCH * HDIM,
                 w2b + (size_t)kn * HDIM * HDIM,
                 b2 + (size_t)src * HDIM,
                 gates + (size_t)kn * BATCH * HDIM,
                 As, Bs);
}

// ---------------------------------------------------------------------------
// cy2 = f2*cx2 + i2*tanh(zg)   (gates: [0]=i2 net1, [1]=f2 net3, [2]=zg net5)
// ---------------------------------------------------------------------------
__launch_bounds__(256)
__global__ void combine(const bf16_t* __restrict__ gates,
                        const float* __restrict__ cx2,
                        float* __restrict__ out)
{
    const size_t N = (size_t)BATCH * HDIM;
    size_t i = ((size_t)blockIdx.x * blockDim.x + threadIdx.x) * 4;
    if (i >= N) return;
    v4bf gi = *(const v4bf*)(gates + 0 * N + i);
    v4bf gf = *(const v4bf*)(gates + 1 * N + i);
    v4bf gz = *(const v4bf*)(gates + 2 * N + i);
    float4 c = *(const float4*)(cx2 + i);
    float4 o;
    o.x = (float)gf[0] * c.x + (float)gi[0] * tanhf((float)gz[0]);
    o.y = (float)gf[1] * c.y + (float)gi[1] * tanhf((float)gz[1]);
    o.z = (float)gf[2] * c.z + (float)gi[2] * tanhf((float)gz[2]);
    o.w = (float)gf[3] * c.w + (float)gi[3] * tanhf((float)gz[3]);
    *(float4*)(out + i) = o;
}

// ---------------------------------------------------------------------------
// Launch
// ---------------------------------------------------------------------------
extern "C" void kernel_launch(void* const* d_in, const int* in_sizes, int n_in,
                              void* d_out, int out_size, void* d_ws, size_t ws_size,
                              hipStream_t stream) {
    const float* hx  = (const float*)d_in[0];
    // d_in[1] = cx1 (dead code in reference)
    const float* cx2 = (const float*)d_in[2];
    const float* W1  = (const float*)d_in[3];
    const float* b1  = (const float*)d_in[4];
    const float* W2  = (const float*)d_in[5];
    const float* b2  = (const float*)d_in[6];

    char* ws = (char*)d_ws;
    size_t off = 0;
    bf16_t* hxb   = (bf16_t*)(ws + off); off += (size_t)BATCH * DIN * sizeof(bf16_t);
    bf16_t* w1b   = (bf16_t*)(ws + off); off += (size_t)3 * HDIM * DIN * sizeof(bf16_t);
    bf16_t* w2b   = (bf16_t*)(ws + off); off += (size_t)3 * HDIM * HDIM * sizeof(bf16_t);
    bf16_t* a1    = (bf16_t*)(ws + off); off += (size_t)3 * BATCH * HDIM * sizeof(bf16_t);
    bf16_t* gates = (bf16_t*)(ws + off); off += (size_t)3 * BATCH * HDIM * sizeof(bf16_t);

    // pack activations and live-net weights to bf16
    {
        int n = BATCH * DIN;
        cvt_f32_bf16<<<n / 8 / 256, 256, 0, stream>>>(hx, hxb, n);
    }
    for (int kn = 0; kn < 3; ++kn) {
        int src = 2 * kn + 1;
        int nw = HDIM * DIN;
        cvt_f32_bf16<<<nw / 8 / 256, 256, 0, stream>>>(
            W1 + (size_t)src * HDIM * DIN, w1b + (size_t)kn * HDIM * DIN, nw);
        cvt_f32_bf16<<<nw / 8 / 256, 256, 0, stream>>>(
            W2 + (size_t)src * HDIM * HDIM, w2b + (size_t)kn * HDIM * HDIM, nw);
    }

    dim3 g1(BATCH / WGM, HDIM / WGN, 3);
    gemm1_tanh<<<g1, 256, 0, stream>>>(hxb, w1b, b1, a1);
    gemm2_gate<<<g1, 256, 0, stream>>>(a1, w2b, b2, gates);

    int nc = BATCH * HDIM / 4;
    combine<<<nc / 256, 256, 0, stream>>>(gates, cx2, (float*)d_out);
}
